// ContextualLSTMCell_2791728742744
// MI455X (gfx1250) — matrix operand order repaired and checked by reading, not compiled, not run on hardware
//
#include <hip/hip_runtime.h>
#include <hip/hip_bf16.h>
#include <cstddef>

// ---------------------------------------------------------------------------
// Types for CDNA5 WMMA
// ---------------------------------------------------------------------------
typedef __attribute__((ext_vector_type(16))) __bf16 v16bf;
typedef __attribute__((ext_vector_type(8)))  __bf16 v8bf;
typedef __attribute__((ext_vector_type(8)))  float  v8f;

static __device__ __forceinline__ unsigned short f2bf(float f) {
    unsigned int u = __builtin_bit_cast(unsigned int, f);
    // round-to-nearest-even bf16 truncation
    unsigned int r = (u + 0x7FFFu + ((u >> 16) & 1u)) >> 16;
    return (unsigned short)r;
}

struct BfPair { v8bf lo, hi; };
static __device__ __forceinline__ v16bf make16(v8bf lo, v8bf hi) {
    BfPair t{lo, hi};
    return __builtin_bit_cast(v16bf, t);
}

// ---------------------------------------------------------------------------
// Problem constants
// ---------------------------------------------------------------------------
#define IDIM   256
#define HDIM   256
#define TDIM   64
#define BATCH  65536
#define KTOT   832          // I + H + H + T
#define KC     32           // K per WMMA step
#define NSTEP  (KTOT / KC)  // 26
#define WELEMS (4 * HDIM * KTOT)   // packed weight elements (bf16)

// ---------------------------------------------------------------------------
// Prep kernel 1: pack 15 weight matrices into bf16 Wpacked[4][256][832]
// gate order: 0=i, 1=f, 2=g (cell cand), 3=o. K blocks: [x | h | c | topic].
// g==2 has zero c-block; g==3 h-block is negated (reference uses -w_ho).
// ---------------------------------------------------------------------------
__global__ __launch_bounds__(256) void pack_weights(
    const float* __restrict__ w_ii, const float* __restrict__ w_hi,
    const float* __restrict__ w_ci, const float* __restrict__ w_bi,
    const float* __restrict__ w_if, const float* __restrict__ w_hf,
    const float* __restrict__ w_cf, const float* __restrict__ w_bf,
    const float* __restrict__ w_ic, const float* __restrict__ w_hc,
    const float* __restrict__ w_bc,
    const float* __restrict__ w_io, const float* __restrict__ w_ho,
    const float* __restrict__ w_co, const float* __restrict__ w_bo,
    unsigned short* __restrict__ wp)
{
    int id = blockIdx.x * blockDim.x + threadIdx.x;
    if (id >= WELEMS) return;
    int g   = id / (HDIM * KTOT);
    int rem = id % (HDIM * KTOT);
    int row = rem / KTOT;
    int k   = rem % KTOT;

    float v = 0.0f;
    if (k < 256) {                       // x block
        const float* w = (g == 0) ? w_ii : (g == 1) ? w_if : (g == 2) ? w_ic : w_io;
        v = w[row * IDIM + k];
    } else if (k < 512) {                // h block
        const float* w = (g == 0) ? w_hi : (g == 1) ? w_hf : (g == 2) ? w_hc : w_ho;
        v = w[row * HDIM + (k - 256)];
        if (g == 3) v = -v;              // output gate: minus sign on hidden term
    } else if (k < 768) {                // c block (zero for cell candidate)
        if (g != 2) {
            const float* w = (g == 0) ? w_ci : (g == 1) ? w_cf : w_co;
            v = w[row * HDIM + (k - 512)];
        }
    } else {                             // topic block
        const float* w = (g == 0) ? w_bi : (g == 1) ? w_bf : (g == 2) ? w_bc : w_bo;
        v = w[row * TDIM + (k - 768)];
    }
    wp[id] = f2bf(v);
}

// ---------------------------------------------------------------------------
// Prep kernel 2: pack biases f32 bp[4][256]
// ---------------------------------------------------------------------------
__global__ __launch_bounds__(256) void pack_bias(
    const float* __restrict__ bi, const float* __restrict__ bff,
    const float* __restrict__ bc, const float* __restrict__ bo,
    float* __restrict__ bp)
{
    int id = blockIdx.x * blockDim.x + threadIdx.x;
    if (id >= 4 * HDIM) return;
    int g = id >> 8, row = id & 255;
    const float* s = (g == 0) ? bi : (g == 1) ? bff : (g == 2) ? bc : bo;
    bp[id] = s[row];
}

// ---------------------------------------------------------------------------
// Main kernel: one workgroup = all 256 H rows x 16 batch columns.
// 8 waves; wave w owns H rows [32w, 32w+32) for all 4 gates.
// Double-buffered LDS staging: stage step s+1 while computing step s
// (one barrier per K-step). Packed f32x2 -> bf16x2 conversion on stage.
// ---------------------------------------------------------------------------
__global__ __launch_bounds__(256) void lstm_main(
    const float* __restrict__ x, const float* __restrict__ h,
    const float* __restrict__ c, const float* __restrict__ topic,
    const unsigned short* __restrict__ wp, const float* __restrict__ bp,
    float* __restrict__ out)
{
    // Double-buffered tile: 2 x 16 batch-cols x 32 K values (bf16),
    // row stride 48 shorts = 96B (mult of 16B -> aligned ds_load_b128).
    __shared__ alignas(16) unsigned short smem[2][16][48];
    __shared__ float sbias[4 * HDIM];

    const int tid  = threadIdx.x;
    const int wave = tid >> 5;
    const int lane = tid & 31;
    const int n    = lane & 15;   // column within 16-wide B tile / A row
    const int hi   = lane >> 4;   // half-wave select
    const size_t b0 = (size_t)blockIdx.x * 16;

    // biases -> LDS (4 KB), once per block
#pragma unroll
    for (int i = 0; i < 4; ++i) sbias[i * 256 + tid] = bp[i * 256 + tid];

    // staging role of this thread: two consecutive K rows of one column
    const int scol  = tid & 15;
    const int skrow = (tid >> 4) << 1;   // 0,2,...,30

    auto stage = [&](int step, int p) {
        const int k0 = step * KC;
        const float* src;
        int krow;
        if (k0 < 256)      { src = x;     krow = k0;       }
        else if (k0 < 512) { src = h;     krow = k0 - 256; }
        else if (k0 < 768) { src = c;     krow = k0 - 512; }
        else               { src = topic; krow = k0 - 768; }
        const size_t base = (size_t)(krow + skrow) * BATCH + b0 + scol;
        float va = src[base];
        float vb = src[base + BATCH];
        __hip_bfloat162 pk = __float22bfloat162_rn(make_float2(va, vb));
        unsigned int u;
        __builtin_memcpy(&u, &pk, sizeof(u));   // bf16x2 -> u32 (byte copy)
        *(unsigned int*)&smem[p][scol][skrow] = u;
    };

    v8f acc[4][2] = {};           // [gate][16-row subtile]

    stage(0, 0);
    __syncthreads();

    for (int step = 0; step < NSTEP; ++step) {
        const int p  = step & 1;
        const int k0 = step * KC;

        // prefetch next tile into the other buffer (no hazard: its previous
        // readers finished before the barrier that ended the last iteration)
        if (step + 1 < NSTEP) stage(step + 1, p ^ 1);

        // B operand: lane holds col n; K = 8*hi..+7 and 16+8*hi..+7
        const v8bf* brow = (const v8bf*)&smem[p][n][hi * 8];
        const v16bf bmat = make16(brow[0], brow[2]);

#pragma unroll
        for (int g = 0; g < 4; ++g) {
#pragma unroll
            for (int s = 0; s < 2; ++s) {
                const int hrow = wave * 32 + s * 16 + n;   // A-matrix row M
                const unsigned short* ap =
                    wp + ((size_t)(g * HDIM + hrow)) * KTOT + k0 + hi * 8;
                v8bf alo = *(const v8bf*)ap;          // K = k0+koff .. +7
                v8bf ahi = *(const v8bf*)(ap + 16);   // K = k0+16+koff ..
                acc[g][s] = __builtin_amdgcn_wmma_f32_16x16x32_bf16(
                    false, make16(alo, ahi), false, bmat,
                    (short)0, acc[g][s], false, false);
            }
        }
        __syncthreads();
    }

    // ------------------ fused gate epilogue ------------------
    const size_t HB = (size_t)HDIM * BATCH;
#pragma unroll
    for (int s = 0; s < 2; ++s) {
#pragma unroll
        for (int r = 0; r < 8; ++r) {
            const int m   = r + hi * 8;                 // C/D layout M index
            const int row = wave * 32 + s * 16 + m;
            const size_t col = b0 + n;

            float pi = acc[0][s][r] + sbias[0 * HDIM + row];
            float pf = acc[1][s][r] + sbias[1 * HDIM + row];
            float pg = acc[2][s][r] + sbias[2 * HDIM + row];
            float po = acc[3][s][r] + sbias[3 * HDIM + row];

            float cv = c[(size_t)row * BATCH + col];

            float si = 1.0f / (1.0f + __expf(-pi));
            float sf = 1.0f / (1.0f + __expf(-pf));
            float gg = tanhf(pg);
            float so = 1.0f / (1.0f + __expf(-po));

            float ccv = sf * cv + si * gg;
            float chv = so * tanhf(ccv);

            out[(size_t)row * BATCH + col]      = chv;  // ch
            out[HB + (size_t)row * BATCH + col] = ccv;  // cc
        }
    }
}

// ---------------------------------------------------------------------------
// Launch
// ---------------------------------------------------------------------------
extern "C" void kernel_launch(void* const* d_in, const int* in_sizes, int n_in,
                              void* d_out, int out_size, void* d_ws, size_t ws_size,
                              hipStream_t stream)
{
    (void)in_sizes; (void)n_in; (void)out_size; (void)ws_size;

    const float* x     = (const float*)d_in[0];
    const float* h     = (const float*)d_in[1];
    const float* c     = (const float*)d_in[2];
    const float* topic = (const float*)d_in[3];

    const float* w_ii = (const float*)d_in[4];
    const float* w_hi = (const float*)d_in[5];
    const float* w_ci = (const float*)d_in[6];
    const float* w_bi = (const float*)d_in[7];
    const float* b_i  = (const float*)d_in[8];

    const float* w_if = (const float*)d_in[9];
    const float* w_hf = (const float*)d_in[10];
    const float* w_cf = (const float*)d_in[11];
    const float* w_bf = (const float*)d_in[12];
    const float* b_f  = (const float*)d_in[13];

    const float* w_ic = (const float*)d_in[14];
    const float* w_hc = (const float*)d_in[15];
    const float* w_bc = (const float*)d_in[16];
    const float* b_c  = (const float*)d_in[17];

    const float* w_io = (const float*)d_in[18];
    const float* w_ho = (const float*)d_in[19];
    const float* w_co = (const float*)d_in[20];
    const float* w_bo = (const float*)d_in[21];
    const float* b_o  = (const float*)d_in[22];

    unsigned short* wp = (unsigned short*)d_ws;
    float* bp = (float*)((char*)d_ws + (size_t)WELEMS * sizeof(unsigned short));

    pack_weights<<<(WELEMS + 255) / 256, 256, 0, stream>>>(
        w_ii, w_hi, w_ci, w_bi,
        w_if, w_hf, w_cf, w_bf,
        w_ic, w_hc, w_bc,
        w_io, w_ho, w_co, w_bo, wp);

    pack_bias<<<4, 256, 0, stream>>>(b_i, b_f, b_c, b_o, bp);

    lstm_main<<<BATCH / 16, 256, 0, stream>>>(
        x, h, c, topic, wp, bp, (float*)d_out);
}
